// SMCN_64244120814291
// MI455X (gfx1250) — matrix operand order repaired and checked by reading, not compiled, not run on hardware
//
#include <hip/hip_runtime.h>

// Problem dims (compile-time constants from the reference)
static constexpr int kT    = 256;
static constexpr int kBS   = 64;
static constexpr int kN    = 128;   // particles
static constexpr int kDin  = 64;
static constexpr int kDout = 16;
static constexpr int kH    = 64;

typedef __attribute__((ext_vector_type(16))) __bf16 v16bf;
typedef __attribute__((ext_vector_type(8)))  float  v8f;

union Frag { v16bf v; unsigned int u[8]; };

// ---------------- deterministic counter-based RNG ----------------
__device__ __forceinline__ unsigned int mix32(unsigned int h) {
    h ^= h >> 16; h *= 0x85EBCA6Bu;
    h ^= h >> 13; h *= 0xC2B2AE35u;
    h ^= h >> 16; return h;
}
__device__ __forceinline__ unsigned int hash4(unsigned int a, unsigned int b,
                                              unsigned int c, unsigned int d) {
    unsigned int h = 0x9E3779B9u;
    h = mix32(h ^ (a * 0x85EBCA77u));
    h = mix32(h ^ (b * 0xC2B2AE3Du));
    h = mix32(h ^ (c * 0x27D4EB2Fu));
    h = mix32(h ^ (d * 0x165667B1u));
    return h;
}
__device__ __forceinline__ float u01(unsigned int h) {
    return (float)(h >> 8) * 0x1p-24f + 0x1p-25f;   // (0,1)
}
// Box-Muller using raw transcendental units:
//   v_log_f32 is log2 -> -2 ln u = -2*ln2 * log2(u)
//   v_cos_f32 input is in revolutions -> cos(2*pi*u) == v_cos(u)
__device__ __forceinline__ float gaussf(unsigned int h1) {
    unsigned int h2 = mix32(h1 ^ 0x68E31DA4u);
    float r = __builtin_sqrtf(-1.38629436112f * __builtin_amdgcn_logf(u01(h1)));
    float c = __builtin_amdgcn_cosf(u01(h2));
    return r * c;
}
// Branch-free tanh: tanh(x) = sign(x) * (1-e)/(1+e), e = exp(-2|x|) via v_exp_f32
__device__ __forceinline__ float tanh_fast(float x) {
    float a = __builtin_fabsf(x);
    float e = __builtin_amdgcn_exp2f(-2.88539008178f * a);  // exp(-2a)
    float t = (1.0f - e) * __builtin_amdgcn_rcpf(1.0f + e);
    return __builtin_copysignf(t, x);
}
// exp(x) via v_exp_f32
__device__ __forceinline__ float exp_fast(float x) {
    return __builtin_amdgcn_exp2f(1.44269504089f * x);
}
// ---------------- bf16 helpers (bit-level, RNE) ----------------
__device__ __forceinline__ unsigned short f2bf(float f) {
    unsigned int x = __float_as_uint(f);
    x += 0x7FFFu + ((x >> 16) & 1u);
    return (unsigned short)(x >> 16);
}
__device__ __forceinline__ float bf2f(unsigned short h) {
    return __uint_as_float(((unsigned int)h) << 16);
}

// A-fragment: 16x32 bf16 tile, row-major source [rows][64], per ISA layout
__device__ __forceinline__ v16bf load_a(const unsigned short* base, int mt, int ks,
                                        int ln, int hl) {
    Frag f;
    const unsigned int* p = (const unsigned int*)(base + (mt * 16 + ln) * 64);
#pragma unroll
    for (int v = 0; v < 8; ++v)
        f.u[v] = p[ks * 16 + (v >> 2) * 8 + hl * 4 + (v & 3)];
    return f.v;
}
// B-fragment: 32x16 bf16 tile from transposed weights W^T[n][k]
__device__ __forceinline__ v16bf load_b(const unsigned short* baseT, int nt, int ks,
                                        int ln, int hl) {
    Frag f;
    const unsigned int* p = (const unsigned int*)(baseT + (nt * 16 + ln) * 64);
#pragma unroll
    for (int v = 0; v < 8; ++v)
        f.u[v] = p[ks * 16 + (v >> 2) * 8 + hl * 4 + (v & 3)];
    return f.v;
}

__global__ __launch_bounds__(256, 1)
void smcn_particle_filter(const float* __restrict__ u, const float* __restrict__ y,
                          const float* __restrict__ W_ih, const float* __restrict__ W_hh,
                          const float* __restrict__ b_ih, const float* __restrict__ b_hh,
                          const float* __restrict__ W1,   const float* __restrict__ b1,
                          const float* __restrict__ W2,   const float* __restrict__ b2,
                          const float* __restrict__ sigx, const float* __restrict__ sigy,
                          float* __restrict__ out) {
    // LDS: whole recurrence stays resident (~61 KB / workgroup)
    __shared__ __align__(16) unsigned short sXa[kN * kDin];      // state (and FFN hidden H)
    __shared__ __align__(16) unsigned short sXb[kN * kDin];      // post-tanh+noise state
    __shared__ __align__(16) unsigned short sWhhT[kDin * kDin];  // [n][k] bf16
    __shared__ __align__(16) unsigned short sW1T[kH * kDin];     // [n][k] bf16
    __shared__ __align__(16) unsigned short sW2T[kDout * kH];    // [n][k] bf16
    __shared__ __align__(16) unsigned short sWih[kDin * kDin];   // [k][n] bf16
    __shared__ float sUvec[kDin], sUproj[kDin], sBias[kDin], sB1[kH];
    __shared__ float sYk[kDout], sB2[kDout], sStdY[kDout], sSigYI[kDout];
    __shared__ float sStdX[kDin];
    __shared__ float sLogw[kN];        // per-particle sum of d^2/sigma_y
    __shared__ float sScanA[kN], sScanB[kN];
    __shared__ float sMin;
    __shared__ int   sIdx[kN];

    const int tid  = threadIdx.x;
    const int lane = tid & 31;
    const int wave = tid >> 5;
    const int hl   = lane >> 4;   // half of wave
    const int ln   = lane & 15;
    const int b    = blockIdx.x;  // one workgroup per batch element

    // ---- one-time weight staging (transpose + bf16) ----
    for (int i = tid; i < kDin * kDin; i += 256) {
        int k = i >> 6, n = i & 63;
        sWhhT[n * 64 + k] = f2bf(W_hh[i]);
        sW1T [n * 64 + k] = f2bf(W1[i]);
        sWih [i]          = f2bf(W_ih[i]);
    }
    for (int i = tid; i < kH * kDout; i += 256) {
        int k = i >> 4, n = i & 15;
        sW2T[n * 64 + k] = f2bf(W2[i]);
    }
    if (tid < kDin) {
        sB1[tid]   = b1[tid];
        sStdX[tid] = __builtin_sqrtf(sigx[tid]);
        sBias[tid] = b_ih[tid] + b_hh[tid];
    }
    if (tid < kDout) {
        float sv = sigy[tid];
        sB2[tid]    = b2[tid];
        sStdY[tid]  = __builtin_sqrtf(sv);
        sSigYI[tid] = 1.0f / sv;
    }
    // x0 ~ N(0,1)
    for (int i = tid; i < kN * kDin; i += 256)
        sXa[i] = f2bf(gaussf(hash4(0xFFFFu, b, i, 0u)));
    __syncthreads();

    for (int t = 0; t < kT; ++t) {
        // ---- stage 1: stage u_t, y_t; prefetch next step's rows ----
        if (tid < kDin)              sUvec[tid]      = u[(t * kBS + b) * kDin + tid];
        else if (tid < kDin + kDout) sYk[tid - kDin] = y[(t * kBS + b) * kDout + (tid - kDin)];
        if (t + 1 < kT && tid == 0) {
            __builtin_prefetch(&u[((t + 1) * kBS + b) * kDin], 0, 0);   // global_prefetch_b8
            __builtin_prefetch(&y[((t + 1) * kBS + b) * kDout], 0, 0);
        }
        __syncthreads();

        // ---- stage 2: uproj = u_t @ W_ih + b_ih + b_hh (shared across particles) ----
        if (tid < kDin) {
            float acc = sBias[tid];
#pragma unroll 8
            for (int k = 0; k < kDin; ++k) acc += sUvec[k] * bf2f(sWih[k * 64 + tid]);
            sUproj[tid] = acc;
        }
        __syncthreads();

        // ---- stage 3: Xb = tanh(Xa @ W_hh + uproj) + std_x * eta  (WMMA) ----
        for (int j = 0; j < 4; ++j) {
            int job = wave * 4 + j;
            int mt = job >> 2, nt = job & 3;
            v8f acc = {};
#pragma unroll
            for (int ks = 0; ks < 2; ++ks) {
                v16bf fa = load_a(sXa, mt, ks, ln, hl);
                v16bf fb = load_b(sWhhT, nt, ks, ln, hl);
                acc = __builtin_amdgcn_wmma_f32_16x16x32_bf16(false, fa, false, fb,
                                                              (short)0, acc, false, false);
            }
#pragma unroll
            for (int v = 0; v < 8; ++v) {
                int row = mt * 16 + hl * 8 + v;
                int col = nt * 16 + ln;
                unsigned int e = (unsigned int)(row * 64 + col);
                float z  = acc[v] + sUproj[col];
                float xv = tanh_fast(z) + sStdX[col] * gaussf(hash4(t, b, e, 1u));
                sXb[row * 64 + col] = f2bf(xv);
            }
        }
        __syncthreads();

        // ---- stage 4: H = relu(Xb @ W1 + b1) -> sXa (old state is dead) ----
        for (int j = 0; j < 4; ++j) {
            int job = wave * 4 + j;
            int mt = job >> 2, nt = job & 3;
            v8f acc = {};
#pragma unroll
            for (int ks = 0; ks < 2; ++ks) {
                v16bf fa = load_a(sXb, mt, ks, ln, hl);
                v16bf fb = load_b(sW1T, nt, ks, ln, hl);
                acc = __builtin_amdgcn_wmma_f32_16x16x32_bf16(false, fa, false, fb,
                                                              (short)0, acc, false, false);
            }
#pragma unroll
            for (int v = 0; v < 8; ++v) {
                int row = mt * 16 + hl * 8 + v;
                int col = nt * 16 + ln;
                float hv = acc[v] + sB1[col];
                sXa[row * 64 + col] = f2bf(hv > 0.0f ? hv : 0.0f);
            }
        }
        __syncthreads();

        // ---- stage 5: yhat = H @ W2 + b2 + std_y * eps; emit preds; row sums ----
        {
            int mt = wave;  // 8 waves <-> 8 M-tiles, single 16-wide N tile
            v8f acc = {};
#pragma unroll
            for (int ks = 0; ks < 2; ++ks) {
                v16bf fa = load_a(sXa, mt, ks, ln, hl);
                v16bf fb = load_b(sW2T, 0, ks, ln, hl);
                acc = __builtin_amdgcn_wmma_f32_16x16x32_bf16(false, fa, false, fb,
                                                              (short)0, acc, false, false);
            }
#pragma unroll
            for (int v = 0; v < 8; ++v) {
                int row = mt * 16 + hl * 8 + v;
                int col = ln;
                unsigned int e = (unsigned int)(row * 16 + col);
                float yh = acc[v] + sB2[col] + sStdY[col] * gaussf(hash4(t, b, e, 2u));
                out[((size_t)(t * kBS + b) * kN + row) * kDout + col] = yh;
                float d = yh - sYk[col];
                float s = d * d * sSigYI[col];
                // reduce over the 16 lanes that share this row (cross-lane permutes)
                s += __shfl_xor(s, 1);
                s += __shfl_xor(s, 2);
                s += __shfl_xor(s, 4);
                s += __shfl_xor(s, 8);
                if (ln == 0) sLogw[row] = s;   // each row written exactly once
            }
        }
        __syncthreads();

        // ---- stage 6: multinomial resampling via inverse-CDF ----
        // 6a: min over sum-squares (= max log-weight) for numerical stability
        if (tid < 32) {
            float m = sLogw[tid];
            m = __builtin_fminf(m, sLogw[tid + 32]);
            m = __builtin_fminf(m, sLogw[tid + 64]);
            m = __builtin_fminf(m, sLogw[tid + 96]);
            m = __builtin_fminf(m, __shfl_xor(m, 16));
            m = __builtin_fminf(m, __shfl_xor(m, 8));
            m = __builtin_fminf(m, __shfl_xor(m, 4));
            m = __builtin_fminf(m, __shfl_xor(m, 2));
            m = __builtin_fminf(m, __shfl_xor(m, 1));
            if (tid == 0) sMin = m;
        }
        __syncthreads();
        // 6b: un-normalized weights (constants of the Gaussian log-pdf cancel)
        if (tid < kN) sScanA[tid] = exp_fast(-0.5f * (sLogw[tid] - sMin));
        __syncthreads();
        // 6c: Hillis-Steele inclusive prefix scan (7 steps, ping-pong)
        {
            float* src = sScanA;
            float* dst = sScanB;
#pragma unroll
            for (int off = 1; off < kN; off <<= 1) {
                if (tid < kN) dst[tid] = (tid >= off) ? src[tid] + src[tid - off] : src[tid];
                __syncthreads();
                float* tmp = src; src = dst; dst = tmp;
            }
            // 6d: draw + branchless binary search
            if (tid < kN) {
                float total = src[kN - 1];
                float r = u01(hash4(t, b, (unsigned int)tid, 3u)) * total;
                int lo = 0;
#pragma unroll
                for (int s = kN / 2; s > 0; s >>= 1) {
                    int mid = lo + s;
                    if (mid <= kN - 1 && src[mid - 1] <= r) lo = mid;
                }
                sIdx[tid] = lo > kN - 1 ? kN - 1 : lo;
            }
        }
        __syncthreads();

        // ---- stage 7: gather resampled particles Xb[idx] -> Xa (next state) ----
        for (int i = tid; i < kN * 32; i += 256) {
            int r = i >> 5, c = i & 31;
            ((unsigned int*)sXa)[r * 32 + c] =
                ((const unsigned int*)sXb)[sIdx[r] * 32 + c];
        }
        __syncthreads();
    }
}

extern "C" void kernel_launch(void* const* d_in, const int* in_sizes, int n_in,
                              void* d_out, int out_size, void* d_ws, size_t ws_size,
                              hipStream_t stream) {
    (void)in_sizes; (void)n_in; (void)out_size; (void)d_ws; (void)ws_size;
    const float* u     = (const float*)d_in[0];
    const float* y     = (const float*)d_in[1];
    const float* W_ih  = (const float*)d_in[2];
    const float* W_hh  = (const float*)d_in[3];
    const float* b_ih  = (const float*)d_in[4];
    const float* b_hh  = (const float*)d_in[5];
    const float* W1    = (const float*)d_in[6];
    const float* b1    = (const float*)d_in[7];
    const float* W2    = (const float*)d_in[8];
    const float* b2    = (const float*)d_in[9];
    const float* sigx  = (const float*)d_in[10];
    const float* sigy  = (const float*)d_in[11];
    float* out = (float*)d_out;

    smcn_particle_filter<<<dim3(kBS), dim3(256), 0, stream>>>(
        u, y, W_ih, W_hh, b_ih, b_hh, W1, b1, W2, b2, sigx, sigy, out);
}
